// MPSClassifier_10737418240127
// MI455X (gfx1250) — matrix operand order repaired and checked by reading, not compiled, not run on hardware
//
#include <hip/hip_runtime.h>

// MPS classifier on MI455X (gfx1250, wave32, WMMA).
// Binary-tree contraction of 32x32 matrices; each product = four
// v_wmma_f32_16x16x32_f16 (K=32 covers the whole reduction). Parity-based
// mixed storage (even index -> row-major for A side, odd index -> col-major
// for B side) makes every fragment load a contiguous b128 access.
// Ping-pong f16 workspace (~154 MB) stays resident in the 192 MB L2.

typedef _Float16 v8h  __attribute__((ext_vector_type(8)));
typedef _Float16 v16h __attribute__((ext_vector_type(16)));
typedef float    v8f  __attribute__((ext_vector_type(8)));

#define SIZE_S   784
#define HALF_S   392
#define BATCH    128
#define NLAB     10
#define MELEMS   1024   // 32*32 f16 per matrix
#define L0_OUT   196

// ---------------- WMMA fragment helpers (ISA 7.12.2 layouts) ----------------

// 16-bit A matrix 16x32 (MxK) from ROW-major 32x32 f16, tile row mo.
// lane: M = mo + (lane&15); VGPR0-3 K = 8h..8h+7, VGPR4-7 K = 16+8h.. (h=lane>>4)
__device__ __forceinline__ v16h load_a_row(const _Float16* M, int lane, int mo) {
    int m  = mo + (lane & 15);
    int h8 = (lane >> 4) * 8;
    union { v16h v; v8h p[2]; } u;
    u.p[0] = *(const v8h*)(M + m * 32 + h8);
    u.p[1] = *(const v8h*)(M + m * 32 + 16 + h8);
    return u.v;
}

// 16-bit B matrix 32x16 (KxN) from COL-major storage (element (k,n) at n*32+k).
// lane: N = no + (lane&15); VGPR0-3 K = 8h..8h+7, VGPR4-7 K = 16+8h.. (h=lane>>4)
__device__ __forceinline__ v16h load_b_col(const _Float16* MT, int lane, int no) {
    int n  = no + (lane & 15);
    int kb = (lane >> 4) * 8;
    union { v16h v; v8h p[2]; } u;
    u.p[0] = *(const v8h*)(MT + n * 32 + kb);
    u.p[1] = *(const v8h*)(MT + n * 32 + 16 + kb);
    return u.v;
}

// f32 16x16 D tile -> f16 ROW-major store (strided b16 stores).
__device__ __forceinline__ void store_d_row(_Float16* out, int lane, int mo, int no, v8f d) {
    int n  = no + (lane & 15);
    int rb = mo + (lane >> 4) * 8;
#pragma unroll
    for (int g = 0; g < 8; ++g) out[(rb + g) * 32 + n] = (_Float16)d[g];
}

// f32 16x16 D tile -> f16 COL-major store (one packed b128 store per tile).
__device__ __forceinline__ void store_d_col(_Float16* out, int lane, int mo, int no, v8f d) {
    int n  = no + (lane & 15);
    int mb = mo + (lane >> 4) * 8;
    v8h p;
#pragma unroll
    for (int g = 0; g < 8; ++g) p[g] = (_Float16)d[g];
    *(v8h*)(out + n * 32 + mb) = p;
}

__device__ __forceinline__ v8f wmma32(v16h a, v16h b) {
    v8f c = {};
    return __builtin_amdgcn_wmma_f32_16x16x32_f16(
        /*neg_a=*/false, a, /*neg_b=*/false, b,
        /*c_mod=*/(short)0, c, /*reuse_a=*/false, /*reuse_b=*/false);
}

__device__ __forceinline__ void store_out(_Float16* o, int lane, bool col,
                                          v8f d00, v8f d01, v8f d10, v8f d11) {
    if (col) {
        store_d_col(o, lane, 0,  0,  d00);
        store_d_col(o, lane, 0,  16, d01);
        store_d_col(o, lane, 16, 0,  d10);
        store_d_col(o, lane, 16, 16, d11);
    } else {
        store_d_row(o, lane, 0,  0,  d00);
        store_d_row(o, lane, 0,  16, d01);
        store_d_row(o, lane, 16, 0,  d10);
        store_d_row(o, lane, 16, 16, d11);
    }
}

// ------------- Level 0: fused embedding + first pair product ---------------
// Each lane builds exactly its fragment elements from core_tensors:
// M = c0 + x*(c1-c0). No LDS, no duplicated loads (K-runs per lane disjoint).

__device__ __forceinline__ v16h make_a_site(const float2* cp, float xv, int lane, int mo) {
    int m  = mo + (lane & 15);
    int h8 = (lane >> 4) * 8;
    v16h a;
#pragma unroll
    for (int q = 0; q < 8; ++q) {
        float2 c = cp[m * 32 + h8 + q];
        a[q] = (_Float16)(c.x + xv * (c.y - c.x));
    }
#pragma unroll
    for (int q = 0; q < 8; ++q) {
        float2 c = cp[m * 32 + 16 + h8 + q];
        a[8 + q] = (_Float16)(c.x + xv * (c.y - c.x));
    }
    return a;
}

__device__ __forceinline__ v16h make_b_site(const float2* cp, float xv, int lane, int no) {
    int n  = no + (lane & 15);
    int kb = (lane >> 4) * 8;
    v16h bv;
#pragma unroll
    for (int i = 0; i < 16; ++i) {
        float2 c = cp[((i < 8) ? (kb + i) : (16 + kb + i - 8)) * 32 + n];   // coalesced across lanes at fixed i
        bv[i] = (_Float16)(c.x + xv * (c.y - c.x));
    }
    return bv;
}

__global__ __launch_bounds__(32) void k_embed_pair(
    const float* __restrict__ x,      // (128, 784)
    const float* __restrict__ core,   // (784, 32, 32, 2)
    _Float16* __restrict__ out)       // [2][196][128][1024]
{
    const int lane = threadIdx.x;
    const int idx  = blockIdx.x;
    const int b    = idx & 127;
    const int rest = idx >> 7;
    const int p    = rest % L0_OUT;
    const int h    = rest / L0_OUT;
    const int s0   = h * HALF_S + 2 * p;
    const int s1   = s0 + 1;

    const float xa = x[b * SIZE_S + s0];
    const float xb = x[b * SIZE_S + s1];
    const float2* cA = (const float2*)(core + (size_t)s0 * 2048);
    const float2* cB = (const float2*)(core + (size_t)s1 * 2048);

    v16h a0 = make_a_site(cA, xa, lane, 0);
    v16h a1 = make_a_site(cA, xa, lane, 16);
    v16h b0 = make_b_site(cB, xb, lane, 0);
    v16h b1 = make_b_site(cB, xb, lane, 16);

    v8f d00 = wmma32(a0, b0);
    v8f d01 = wmma32(a0, b1);
    v8f d10 = wmma32(a1, b0);
    v8f d11 = wmma32(a1, b1);

    _Float16* o = out + (((size_t)(h * L0_OUT + p)) * BATCH + b) * MELEMS;
    store_out(o, lane, (p & 1) != 0, d00, d01, d10, d11);
}

// ------------- Generic tree level: pairwise products + lone carry ----------
// in[2j] row-major (A side), in[2j+1] col-major (B side). Output format by
// next-level role: row if j even, col if j odd; final level: left row, right col.
__global__ __launch_bounds__(32) void k_level(
    const _Float16* __restrict__ in, _Float16* __restrict__ out,
    int n_in, int n_out, int final_lvl)
{
    const int lane = threadIdx.x;
    const int idx  = blockIdx.x;
    const int b    = idx & 127;
    const int rest = idx >> 7;
    const int j    = rest % n_out;
    const int h    = rest / n_out;
    const int n_pairs = n_in >> 1;

    _Float16* o = out + (((size_t)h * n_out + j) * BATCH + b) * MELEMS;

    if (j >= n_pairs) {  // lone carry (wave-uniform branch); source is row-major
        const _Float16* src = in + (((size_t)h * n_in + (n_in - 1)) * BATCH + b) * MELEMS;
        if ((n_pairs & 1) == 0) {            // dest row-major: straight copy
            const v8h* s8 = (const v8h*)src;
            v8h* o8 = (v8h*)o;
#pragma unroll
            for (int t = 0; t < 4; ++t) o8[t * 32 + lane] = s8[t * 32 + lane];
        } else {                             // dest col-major: transpose copy
            union { v8h v[4]; _Float16 e[32]; } r;
            const v8h* s8 = (const v8h*)(src + lane * 32);
#pragma unroll
            for (int t = 0; t < 4; ++t) r.v[t] = s8[t];
#pragma unroll
            for (int q = 0; q < 32; ++q) o[q * 32 + lane] = r.e[q];
        }
        return;
    }

    const _Float16* A  = in + (((size_t)h * n_in + 2 * j    ) * BATCH + b) * MELEMS;
    const _Float16* BT = in + (((size_t)h * n_in + 2 * j + 1) * BATCH + b) * MELEMS;

    v16h a0 = load_a_row(A,  lane, 0);
    v16h a1 = load_a_row(A,  lane, 16);
    v16h b0 = load_b_col(BT, lane, 0);
    v16h b1 = load_b_col(BT, lane, 16);

    v8f d00 = wmma32(a0, b0);
    v8f d01 = wmma32(a0, b1);
    v8f d10 = wmma32(a1, b0);
    v8f d11 = wmma32(a1, b1);

    const bool col = final_lvl ? (h == 1) : ((j & 1) != 0);
    store_out(o, lane, col, d00, d01, d10, d11);
}

// ------------- Final scoring on WMMA: scores[b,l] = tr(left^T? ...) --------
// Q_l = label[:,:,l] (32x32, jxk) x right (32x32, kxi)  -> 4 WMMAs
// scores[b,l] = sum_{i,j} left[i,j] * Q_l[j,i]          -> fused trace
__device__ __forceinline__ v16h make_a_label(const float* label, int l, int lane, int jo) {
    int m  = jo + (lane & 15);
    int h8 = (lane >> 4) * 8;
    v16h a;
#pragma unroll
    for (int q = 0; q < 8; ++q)
        a[q] = (_Float16)label[(size_t)(m * 32 + h8 + q) * NLAB + l];
#pragma unroll
    for (int q = 0; q < 8; ++q)
        a[8 + q] = (_Float16)label[(size_t)(m * 32 + 16 + h8 + q) * NLAB + l];
    return a;
}

__device__ __forceinline__ float tracc(v8f q, const _Float16* left, int lane, int jo, int io) {
    // lane element g of Q tile (jo,io): j = jo + 8*(lane>>4) + g, i = io + (lane&15)
    const _Float16* lp = left + (io + (lane & 15)) * 32 + jo + (lane >> 4) * 8;
    v8h lv = *(const v8h*)lp;
    float s = 0.f;
#pragma unroll
    for (int g = 0; g < 8; ++g) s += q[g] * (float)lv[g];
    return s;
}

__global__ __launch_bounds__(32) void k_score(
    const _Float16* __restrict__ mats,  // [2][1][128][1024]: left row-major, right col-major
    const float* __restrict__ label,    // (32, 32, 10)
    float* __restrict__ out)            // (128, 10)
{
    const int lane = threadIdx.x;
    const int l = blockIdx.x % NLAB;
    const int b = blockIdx.x / NLAB;
    const _Float16* left   = mats + (size_t)b * MELEMS;            // (i,j) at i*32+j
    const _Float16* rightT = mats + (size_t)(BATCH + b) * MELEMS;  // (k,i) at i*32+k

    v16h bb0 = load_b_col(rightT, lane, 0);
    v16h bb1 = load_b_col(rightT, lane, 16);
    v16h aa0 = make_a_label(label, l, lane, 0);
    v16h aa1 = make_a_label(label, l, lane, 16);

    v8f q00 = wmma32(aa0, bb0);
    v8f q01 = wmma32(aa0, bb1);
    v8f q10 = wmma32(aa1, bb0);
    v8f q11 = wmma32(aa1, bb1);

    float acc = tracc(q00, left, lane, 0,  0)
              + tracc(q01, left, lane, 0,  16)
              + tracc(q10, left, lane, 16, 0)
              + tracc(q11, left, lane, 16, 16);

#pragma unroll
    for (int off = 16; off > 0; off >>= 1) acc += __shfl_xor(acc, off, 32);
    if (lane == 0) out[b * NLAB + l] = acc;
}

// ---------------------------------------------------------------------------
extern "C" void kernel_launch(void* const* d_in, const int* in_sizes, int n_in,
                              void* d_out, int out_size, void* d_ws, size_t ws_size,
                              hipStream_t stream) {
    const float* x     = (const float*)d_in[0];   // (128, 784)
    const float* core  = (const float*)d_in[1];   // (784, 32, 32, 2)
    const float* label = (const float*)d_in[2];   // (32, 32, 10)
    float* out = (float*)d_out;                   // (128, 10)

    _Float16* bufA = (_Float16*)d_ws;                                  // up to 2*196*128 mats
    _Float16* bufB = bufA + (size_t)2 * L0_OUT * BATCH * MELEMS;       // up to 2*98*128 mats

    // Level 0: 392 -> 196 per half (fused embedding + pair product)
    k_embed_pair<<<dim3(2 * L0_OUT * BATCH), dim3(32), 0, stream>>>(x, core, bufA);

    // Tree levels: 196 -> 98 -> 49 -> 25 -> 13 -> 7 -> 4 -> 2 -> 1
    int n = L0_OUT;
    _Float16* cur = bufA;
    _Float16* nxt = bufB;
    while (n > 1) {
        const int n_out = (n >> 1) + (n & 1);
        const int fin = (n_out == 1) ? 1 : 0;
        k_level<<<dim3(2 * n_out * BATCH), dim3(32), 0, stream>>>(cur, nxt, n, n_out, fin);
        _Float16* tswap = cur; cur = nxt; nxt = tswap;
        n = n_out;
    }
    // cur == bufA: [2][1][128] final matrices (left row-major, right col-major).

    k_score<<<dim3(BATCH * NLAB), dim3(32), 0, stream>>>(cur, label, out);
}
